// PrimaryKeyGAT_13048110645454
// MI455X (gfx1250) — matrix operand order repaired and checked by reading, not compiled, not run on hardware
//
#include <hip/hip_runtime.h>
#include <stdint.h>

// ---------------------------------------------------------------------------
// Types for CDNA5 WMMA (wave32): A/B = 16 bf16 per lane, C/D = 8 f32 per lane
// ---------------------------------------------------------------------------
typedef __attribute__((ext_vector_type(16))) __bf16 v16bf;
typedef __attribute__((ext_vector_type(8)))  float  v8f;

#define NEG_SLOPE 0.2f

// float -> bf16 (round-to-nearest-even), stored as u16 to avoid relying on
// scalar __bf16 conversion support.
__device__ __forceinline__ unsigned short f2bf(float f) {
  unsigned u = __float_as_uint(f);
  u += 0x7FFFu + ((u >> 16) & 1u);
  return (unsigned short)(u >> 16);
}

// Order-preserving float <-> uint mapping (for atomicMax-based segment max).
__device__ __forceinline__ unsigned fkey(float f) {
  unsigned u = __float_as_uint(f);
  return (u & 0x80000000u) ? ~u : (u | 0x80000000u);
}
__device__ __forceinline__ float funkey(unsigned k) {
  unsigned u = (k & 0x80000000u) ? (k & 0x7FFFFFFFu) : ~k;
  return __uint_as_float(u);
}
#define KEY_NEG_INF 0x007FFFFFu   // fkey(-inf)

// ---------------------------------------------------------------------------
// Fill
// ---------------------------------------------------------------------------
__global__ void k_fill_u32(unsigned* __restrict__ p, unsigned v, long long n) {
  long long i = (long long)blockIdx.x * blockDim.x + threadIdx.x;
  if (i < n) p[i] = v;
}

// ---------------------------------------------------------------------------
// Pack A (row-major f32 [M,K]) into WMMA A-fragment bf16 layout.
// Per (mtile,ktile): 32 lanes x 16 values contiguous (1KB block).
// A 16-bit 16x32 layout: lane l -> M = l%16 ; element i -> VGPR v=i/2:
//   kbase = (v<4 ? 2v : 16+2(v-4)) + (l/16)*8 ; k = kbase + (i&1)
// ---------------------------------------------------------------------------
__device__ __forceinline__ void a_frag_coords(long long idx, int ktiles,
                                              long long* row, int* k) {
  int i    = (int)(idx & 15);
  int lane = (int)((idx >> 4) & 31);
  long long tile = idx >> 9;
  int ktile = (int)(tile % ktiles);
  long long mtile = tile / ktiles;
  *row = mtile * 16 + (lane & 15);
  int v = i >> 1;
  int kbase = (v < 4 ? v * 2 : 16 + (v - 4) * 2) + (lane >> 4) * 8;
  *k = ktile * 32 + kbase + (i & 1);
}

__global__ void k_pack_a(const float* __restrict__ src, unsigned short* __restrict__ dst,
                         int N, int K, int Mtiles) {
  int ktiles = K >> 5;
  long long total = (long long)Mtiles * ktiles * 512;
  long long idx = (long long)blockIdx.x * blockDim.x + threadIdx.x;
  if (idx >= total) return;
  long long row; int k;
  a_frag_coords(idx, ktiles, &row, &k);
  float val = (row < N) ? src[row * (long long)K + k] : 0.f;
  dst[idx] = f2bf(val);
}

// Pack B (row-major f32 [K,Nc]) into WMMA B-fragment bf16 layout.
// B 16-bit 32x16 layout: lane l -> N = l%16 ; element i -> k = (i<8 ? i : 16+(i-8)) + (l/16)*8.
// Tile order: tile = ntile*ktiles + ktile (inner-k contiguous per ntile).
__global__ void k_pack_b(const float* __restrict__ src, unsigned short* __restrict__ dst,
                         int K, int Nc) {
  int ktiles = K >> 5;
  long long total = (long long)K * Nc;
  long long idx = (long long)blockIdx.x * blockDim.x + threadIdx.x;
  if (idx >= total) return;
  int i    = (int)(idx & 15);
  int lane = (int)((idx >> 4) & 31);
  long long tile = idx >> 9;
  int ktile = (int)(tile % ktiles);
  int ntile = (int)(tile / ktiles);
  int n = ntile * 16 + (lane & 15);
  int k = ktile * 32 + (i & 8) * 2 + (lane >> 4) * 8 + (i & 7);
  dst[idx] = f2bf(src[(long long)k * Nc + n]);
}

// ---------------------------------------------------------------------------
// WMMA GEMM: C[M,Nc] f32 = A[M,K] bf16 * B[K,Nc] bf16 (fragment-packed inputs)
// Block = 128 threads (4 waves); each wave computes a 16(M) x 64(N) tile,
// reusing one A fragment across 4 B tiles. K unrolled at compile time.
// ---------------------------------------------------------------------------
template <int KTILES>
__global__ void k_gemm_wmma(const unsigned short* __restrict__ aP,
                            const unsigned short* __restrict__ bP,
                            float* __restrict__ C, int N, int Nc, int Mtiles) {
  int wave = threadIdx.x >> 5;
  int lane = threadIdx.x & 31;
  int mtile = blockIdx.x * 4 + wave;
  if (mtile >= Mtiles) return;
  int ng = blockIdx.y;  // group of 4 ntiles (64 columns)

  v8f acc[4] = {};
  const unsigned short* aBase = aP + ((long long)mtile * KTILES) * 512 + lane * 16;

#pragma unroll
  for (int kt = 0; kt < KTILES; ++kt) {
    v16bf a = *(const v16bf*)(aBase + kt * 512);
#pragma unroll
    for (int j = 0; j < 4; ++j) {
      const unsigned short* bq =
          bP + (((long long)(ng * 4 + j) * KTILES + kt) * 512) + lane * 16;
      v16bf b = *(const v16bf*)bq;
      acc[j] = __builtin_amdgcn_wmma_f32_16x16x32_bf16(
          false, a, false, b, (short)0, acc[j], false, false);
    }
  }

  // C/D layout: lane l holds col = l%16; VGPR v holds row = v + 8*(l/16).
  int rbase = mtile * 16 + (lane >> 4) * 8;
  int col0  = ng * 64 + (lane & 15);
#pragma unroll
  for (int j = 0; j < 4; ++j) {
    int col = col0 + j * 16;
#pragma unroll
    for (int v = 0; v < 8; ++v) {
      int row = rbase + v;
      if (row < N) C[(long long)row * Nc + col] = acc[j][v];
    }
  }
}

// ---------------------------------------------------------------------------
// Attention scores: a_src[n,h] = <h[n,h,:], att_src[h,:]>, ditto a_dst.
// One wave per (node, head); shfl_xor tree reduction.
// ---------------------------------------------------------------------------
__global__ void k_attn(const float* __restrict__ h, const float* __restrict__ att_s,
                       const float* __restrict__ att_d, float* __restrict__ oas,
                       float* __restrict__ oad, int N, int H, int C) {
  int gw = blockIdx.x * (blockDim.x >> 5) + (threadIdx.x >> 5);
  int lane = threadIdx.x & 31;
  if (gw >= N * H) return;
  int n = gw / H, hh = gw - n * H;
  const float* hp = h + ((long long)n * H + hh) * C;
  float s0 = 0.f, s1 = 0.f;
  for (int c = lane; c < C; c += 32) {
    float v = hp[c];
    s0 += v * att_s[hh * C + c];
    s1 += v * att_d[hh * C + c];
  }
#pragma unroll
  for (int off = 16; off; off >>= 1) {
    s0 += __shfl_xor(s0, off, 32);
    s1 += __shfl_xor(s1, off, 32);
  }
  if (lane == 0) { oas[gw] = s0; oad[gw] = s1; }
}

// ---------------------------------------------------------------------------
// Edge pass A: logits = leaky_relu(a_src[src] + a_dst[dst]); segment max via
// order-preserving-uint atomicMax.
// ---------------------------------------------------------------------------
__global__ void k_edge_lmax(const int* __restrict__ srcI, const int* __restrict__ dstI,
                            const float* __restrict__ as, const float* __restrict__ ad,
                            float* __restrict__ elog, unsigned* __restrict__ mkey,
                            int E, int Etot, int H) {
  long long t = (long long)blockIdx.x * blockDim.x + threadIdx.x;
  if (t >= (long long)Etot * H) return;
  int e = (int)(t / H), hh = (int)(t % H);
  int s = (e < E) ? srcI[e] : (e - E);
  int d = (e < E) ? dstI[e] : (e - E);
  float l = as[s * H + hh] + ad[d * H + hh];
  l = (l > 0.f) ? l : NEG_SLOPE * l;
  elog[t] = l;
  atomicMax(mkey + d * H + hh, fkey(l));
}

// Edge pass B: e = exp(logit - m[dst]); segment sum via atomicAdd; store e.
__global__ void k_edge_expsum(const int* __restrict__ dstI, float* __restrict__ elog,
                              const unsigned* __restrict__ mkey, float* __restrict__ ssum,
                              int E, int Etot, int H) {
  long long t = (long long)blockIdx.x * blockDim.x + threadIdx.x;
  if (t >= (long long)Etot * H) return;
  int e = (int)(t / H), hh = (int)(t % H);
  int d = (e < E) ? dstI[e] : (e - E);
  float m = funkey(mkey[d * H + hh]);
  float ex = expf(elog[t] - m);
  elog[t] = ex;
  atomicAdd(ssum + d * H + hh, ex);
}

// Edge pass C: out[dst,h,:] += alpha * h[src,h,:]. One wave per (edge, head),
// VPT = C/32 contiguous floats per lane (vectorized reads, f32 atomic adds).
template <int VPT>
__global__ void k_edge_agg(const int* __restrict__ srcI, const int* __restrict__ dstI,
                           const float* __restrict__ eexp, const float* __restrict__ ssum,
                           const float* __restrict__ hsrc, float* __restrict__ out,
                           int E, int Etot, int H, int C) {
  long long gw = (long long)blockIdx.x * (blockDim.x >> 5) + (threadIdx.x >> 5);
  if (gw >= (long long)Etot * H) return;
  int lane = threadIdx.x & 31;
  int e = (int)(gw / H), hh = (int)(gw % H);
  int s = (e < E) ? srcI[e] : (e - E);
  int d = (e < E) ? dstI[e] : (e - E);
  float alpha = eexp[gw] / (ssum[d * H + hh] + 1e-16f);
  const float* hp = hsrc + ((long long)s * H + hh) * C + lane * VPT;
  float* op = out + ((long long)d * H + hh) * C + lane * VPT;
#pragma unroll
  for (int v = 0; v < VPT; ++v) atomicAdd(op + v, alpha * hp[v]);
}

// ---------------------------------------------------------------------------
// h2 = elu(out1 + b1), written straight into WMMA A-fragment bf16 layout.
// ---------------------------------------------------------------------------
__global__ void k_bias_elu_pack(const float* __restrict__ src, const float* __restrict__ bias,
                                unsigned short* __restrict__ dst, int N, int K, int Mtiles) {
  int ktiles = K >> 5;
  long long total = (long long)Mtiles * ktiles * 512;
  long long idx = (long long)blockIdx.x * blockDim.x + threadIdx.x;
  if (idx >= total) return;
  long long row; int k;
  a_frag_coords(idx, ktiles, &row, &k);
  float val = 0.f;
  if (row < N) {
    float t = src[row * (long long)K + k] + bias[k];
    val = (t > 0.f) ? t : (expf(t) - 1.f);
  }
  dst[idx] = f2bf(val);
}

// Final: out = log_softmax(acc + b, axis=1), row width 64. One wave per row.
__global__ void k_logsoftmax(const float* __restrict__ acc, const float* __restrict__ b,
                             float* __restrict__ out, int N) {
  int gw = blockIdx.x * (blockDim.x >> 5) + (threadIdx.x >> 5);
  int lane = threadIdx.x & 31;
  if (gw >= N) return;
  const float* p = acc + (long long)gw * 64;
  float v0 = p[lane] + b[lane];
  float v1 = p[lane + 32] + b[lane + 32];
  float m = fmaxf(v0, v1);
#pragma unroll
  for (int off = 16; off; off >>= 1) m = fmaxf(m, __shfl_xor(m, off, 32));
  float s = expf(v0 - m) + expf(v1 - m);
#pragma unroll
  for (int off = 16; off; off >>= 1) s += __shfl_xor(s, off, 32);
  float lse = m + logf(s);
  float* o = out + (long long)gw * 64;
  o[lane]      = v0 - lse;
  o[lane + 32] = v1 - lse;
}

// ---------------------------------------------------------------------------
// Launch
// ---------------------------------------------------------------------------
static inline int cdiv_ll(long long a, long long b) { return (int)((a + b - 1) / b); }

extern "C" void kernel_launch(void* const* d_in, const int* in_sizes, int n_in,
                              void* d_out, int out_size, void* d_ws, size_t ws_size,
                              hipStream_t stream) {
  (void)n_in; (void)out_size; (void)ws_size;
  const float* x   = (const float*)d_in[0];
  const int*   ei  = (const int*)  d_in[1];
  const float* W1  = (const float*)d_in[2];
  const float* as1 = (const float*)d_in[3];
  const float* ad1 = (const float*)d_in[4];
  const float* b1  = (const float*)d_in[5];
  const float* W2  = (const float*)d_in[6];
  const float* as2 = (const float*)d_in[7];
  const float* ad2 = (const float*)d_in[8];
  const float* b2  = (const float*)d_in[9];
  float* out = (float*)d_out;

  const int FIN = 256, H1 = 2, C1 = 128, HID = 256, C2 = 64;
  const int N = in_sizes[0] / FIN;
  const int E = in_sizes[1] / 2;
  const int Etot = E + N;
  const int* srcI = ei;       // edge_index[0]
  const int* dstI = ei + E;   // edge_index[1]
  const int Mtiles = (N + 15) / 16;

  // Workspace partition (regions reused with stream-ordered lifetimes).
  char* p = (char*)d_ws;
  auto alloc = [&](size_t bytes) -> char* {
    char* r = p; p += (bytes + 255) & ~(size_t)255; return r;
  };
  unsigned short* aPack = (unsigned short*)alloc((size_t)Mtiles * 16 * FIN * 2); // x_bf16 -> h2_bf16
  unsigned short* w1P   = (unsigned short*)alloc((size_t)FIN * HID * 2);
  unsigned short* w2P   = (unsigned short*)alloc((size_t)HID * C2 * 2);
  float* h1   = (float*)alloc((size_t)N * HID * 4);   // layer-1 h -> layer-2 hh
  float* out1 = (float*)alloc((size_t)N * HID * 4);   // layer-1 aggregate -> layer-2 aggregate
  float* scal = (float*)alloc((size_t)N * 8 * 4);     // per-node scalars
  float* elog = (float*)alloc((size_t)Etot * H1 * 4); // per-edge logits/exp

  // layer-1 scalar slots
  float*    aS1 = scal;
  float*    aD1 = scal + 2 * (size_t)N;
  unsigned* m1  = (unsigned*)(scal + 4 * (size_t)N);
  float*    s1  = scal + 6 * (size_t)N;
  // layer-2 scalar slots (layer-1 slots dead by the time these are written)
  float*    aS2 = scal;
  float*    aD2 = scal + (size_t)N;
  unsigned* m2  = (unsigned*)(scal + 4 * (size_t)N);
  float*    s2  = scal + 6 * (size_t)N;
  float* hh   = h1;    // N x 64
  float* out2 = out1;  // N x 64

  const dim3 B256(256);
  const long long aTot = (long long)Mtiles * 16 * FIN;

  // ---- Layer 1 ----
  k_fill_u32<<<cdiv_ll((long long)N * H1, 256), B256, 0, stream>>>(m1, KEY_NEG_INF, (long long)N * H1);
  k_fill_u32<<<cdiv_ll((long long)N * H1, 256), B256, 0, stream>>>((unsigned*)s1, 0u, (long long)N * H1);
  k_fill_u32<<<cdiv_ll((long long)N * HID, 256), B256, 0, stream>>>((unsigned*)out1, 0u, (long long)N * HID);

  k_pack_a<<<cdiv_ll(aTot, 256), B256, 0, stream>>>(x, aPack, N, FIN, Mtiles);
  k_pack_b<<<cdiv_ll((long long)FIN * HID, 256), B256, 0, stream>>>(W1, w1P, FIN, HID);
  k_pack_b<<<cdiv_ll((long long)HID * C2, 256), B256, 0, stream>>>(W2, w2P, HID, C2);

  dim3 g1(cdiv_ll(Mtiles, 4), HID / 64);
  k_gemm_wmma<8><<<g1, dim3(128), 0, stream>>>(aPack, w1P, h1, N, HID, Mtiles);

  k_attn<<<cdiv_ll((long long)N * H1, 8), B256, 0, stream>>>(h1, as1, ad1, aS1, aD1, N, H1, C1);

  const long long eh1 = (long long)Etot * H1;
  k_edge_lmax  <<<cdiv_ll(eh1, 256), B256, 0, stream>>>(srcI, dstI, aS1, aD1, elog, m1, E, Etot, H1);
  k_edge_expsum<<<cdiv_ll(eh1, 256), B256, 0, stream>>>(dstI, elog, m1, s1, E, Etot, H1);
  k_edge_agg<4><<<cdiv_ll(eh1, 8), B256, 0, stream>>>(srcI, dstI, elog, s1, h1, out1, E, Etot, H1, C1);

  // ---- Between layers: h2 = elu(out1 + b1), packed bf16 for GEMM2 ----
  k_bias_elu_pack<<<cdiv_ll(aTot, 256), B256, 0, stream>>>(out1, b1, aPack, N, HID, Mtiles);

  // ---- Layer 2 ----
  k_fill_u32<<<cdiv_ll((long long)N, 256), B256, 0, stream>>>(m2, KEY_NEG_INF, (long long)N);
  k_fill_u32<<<cdiv_ll((long long)N, 256), B256, 0, stream>>>((unsigned*)s2, 0u, (long long)N);
  k_fill_u32<<<cdiv_ll((long long)N * C2, 256), B256, 0, stream>>>((unsigned*)out2, 0u, (long long)N * C2);

  dim3 g2(cdiv_ll(Mtiles, 4), C2 / 64);
  k_gemm_wmma<8><<<g2, dim3(128), 0, stream>>>(aPack, w2P, hh, N, C2, Mtiles);

  k_attn<<<cdiv_ll((long long)N, 8), B256, 0, stream>>>(hh, as2, ad2, aS2, aD2, N, 1, C2);

  const long long eh2 = (long long)Etot;
  k_edge_lmax  <<<cdiv_ll(eh2, 256), B256, 0, stream>>>(srcI, dstI, aS2, aD2, elog, m2, E, Etot, 1);
  k_edge_expsum<<<cdiv_ll(eh2, 256), B256, 0, stream>>>(dstI, elog, m2, s2, E, Etot, 1);
  k_edge_agg<2><<<cdiv_ll(eh2, 8), B256, 0, stream>>>(srcI, dstI, elog, s2, hh, out2, E, Etot, 1, C2);

  // ---- Final bias + log_softmax ----
  k_logsoftmax<<<cdiv_ll((long long)N, 8), B256, 0, stream>>>(out2, b2, out, N);
}
